// FastAttentionBlock_16286515986679
// MI455X (gfx1250) — matrix operand order repaired and checked by reading, not compiled, not run on hardware
//
#include <hip/hip_runtime.h>

#define HW 16384
#define DEV __device__ __forceinline__

typedef __bf16 bf16_t;
typedef __attribute__((ext_vector_type(16))) __bf16 v16bf;
typedef __attribute__((ext_vector_type(8)))  float  v8f;

union FragBF { v16bf v; bf16_t h[16]; uint4 q[2]; };
union PackBF { bf16_t h[2]; unsigned u; };

DEV v8f wmma_bf16(v16bf a, v16bf b, v8f c) {
  return __builtin_amdgcn_wmma_f32_16x16x32_bf16(false, a, false, b, (short)0, c,
                                                 false, false);
}

// A fragment (16x32 bf16) from row-major bf16 source, K contiguous.
// Layout used here: lanes 0-15 hold M=lane, K=0..15; lanes 16-31 K=16..31.
// Two 16B vector loads per lane.
DEV v16bf a_frag_bf(const bf16_t* p0, int ld, int lane) {
  const bf16_t* p = p0 + (long)(lane & 15) * ld + ((lane >> 4) << 4);
  FragBF f;
  f.q[0] = *(const uint4*)(p);
  f.q[1] = *(const uint4*)(p + 8);
  return f.v;
}

// A fragment from fp32 row-major source (used only for tiny simT operand).
DEV v16bf a_frag_f32(const float* base, int ld, int lane) {
  FragBF f;
  const float* p = base + (long)(lane & 15) * ld + ((lane >> 4) << 4);
#pragma unroll
  for (int i = 0; i < 8; ++i) {
    f.h[i]     = (bf16_t)p[i];
    f.h[8 + i] = (bf16_t)p[8 + i];
  }
  return f.v;
}

// B fragment (32x16 bf16) from LDS tile stored [n][k], row stride 40 (80B: 16B
// aligned, 20-bank stride -> conflict-free). Lanes 0-15: col n=lane, K=0..15;
// lanes 16-31: K=16..31. Two ds_load_b128 per lane.
#define BT_LD 40
DEV v16bf b_frag_lds(const bf16_t* lds, int lane) {
  const bf16_t* p = lds + (lane & 15) * BT_LD + ((lane >> 4) << 4);
  FragBF f;
  f.q[0] = *(const uint4*)(p);
  f.q[1] = *(const uint4*)(p + 8);
  return f.v;
}

// ---------------- weight pre-conversion (fp32 -> bf16, done once, tiny) -------
__global__ void k_f2b(const float* __restrict__ src, bf16_t* __restrict__ dst, int n) {
  int t = blockIdx.x * blockDim.x + threadIdx.x;
  if (t < n) dst[t] = (bf16_t)src[t];
}
// sW [o][ci][ky][kx] -> tap-major bf16 [r][o][ci] so conv A-fragments are contiguous.
__global__ void k_cvt_sw(const float* __restrict__ sW, bf16_t* __restrict__ dst) {
  int t = blockIdx.x * blockDim.x + threadIdx.x;
  if (t >= 9 * 128 * 256) return;
  int r = t >> 15, rem = t & 32767;
  int o = rem >> 8, ci = rem & 255;
  dst[t] = (bf16_t)sW[(o * 256 + ci) * 9 + r];
}

// ---------------- 1x1 conv projections (Q/K/V) as WMMA GEMM ----------------
__global__ void k_proj(const float* __restrict__ x, const bf16_t* __restrict__ W,
                       const float* __restrict__ scale, const float* __restrict__ shift,
                       bf16_t* __restrict__ out, int M, int relu) {
  __shared__ __align__(16) bf16_t bt[16 * BT_LD];
  const int lane = threadIdx.x & 31;
  const int wave = threadIdx.x >> 5;
  const int nw   = blockDim.x >> 5;
  const int b  = blockIdx.x >> 10;          // 1024 pixel tiles per batch
  const int n0 = (blockIdx.x & 1023) << 4;
  const int m0 = (blockIdx.y * nw + wave) << 4;
  const float* xb = x + (long)b * 256 * HW;

  v8f acc = {};
  for (int k0 = 0; k0 < 256; k0 += 32) {
    __syncthreads();
    for (int i = threadIdx.x; i < 256; i += blockDim.x) {
      int j = i & 15, kp = i >> 4;             // consecutive threads -> coalesced n
      PackBF pk;
      pk.h[0] = (bf16_t)xb[(long)(k0 + 2 * kp) * HW + n0 + j];
      pk.h[1] = (bf16_t)xb[(long)(k0 + 2 * kp + 1) * HW + n0 + j];
      *(unsigned*)&bt[j * BT_LD + 2 * kp] = pk.u;
    }
    if (k0 < 224) __builtin_prefetch(xb + (long)(k0 + 32) * HW + n0, 0, 0);
    __syncthreads();
    v16bf a = a_frag_bf(W + (long)m0 * 256 + k0, 256, lane);
    acc = wmma_bf16(a, b_frag_lds(bt, lane), acc);
  }
  const int n  = n0 + (lane & 15);
  const int mh = (lane >> 4) << 3;
  bf16_t* ob = out + (long)b * M * HW;
#pragma unroll
  for (int r = 0; r < 8; ++r) {
    int m = m0 + r + mh;
    float vv = acc[r] * scale[m] + shift[m];
    if (relu) vv = fmaxf(vv, 0.f);
    ob[(long)m * HW + n] = (bf16_t)vv;
  }
}

// ---------------- per-pixel inverse L2 norms of q and k ----------------
__global__ void k_norm(const bf16_t* __restrict__ Q, const bf16_t* __restrict__ K,
                       float* __restrict__ invq, float* __restrict__ invk) {
  int t = blockIdx.x * blockDim.x + threadIdx.x;       // 0..65535
  int b = t >> 14, n = t & 16383;
  const bf16_t* qb = Q + (long)b * 64 * HW + n;
  const bf16_t* kb = K + (long)b * 64 * HW + n;
  float sq = 0.f, sk = 0.f;
#pragma unroll 4
  for (int c = 0; c < 64; ++c) {
    float a = (float)qb[(long)c * HW]; sq += a * a;
    float e = (float)kb[(long)c * HW]; sk += e * e;
  }
  invq[t] = 1.f / fmaxf(sqrtf(sq), 1e-12f);
  invk[t] = 1.f / fmaxf(sqrtf(sk), 1e-12f);
}

__global__ void k_zero(float* __restrict__ p, int n) {
  int t = blockIdx.x * blockDim.x + threadIdx.x;
  if (t < n) p[t] = 0.f;
}

// ---------------- sim = k_hat . v'^T  (split-K over HW, atomic f32) ----------------
// simT[b][j][c] += sum_n k[c][n] * invk[n] * v[j][n]
__global__ void k_sim(const bf16_t* __restrict__ Kb, const bf16_t* __restrict__ V,
                      const float* __restrict__ invk, float* __restrict__ simT) {
  int id = blockIdx.x;
  int chunk = id & 7;  id >>= 3;   // 8 HW chunks of 2048
  int jt = id & 15;    id >>= 4;   // 16 tiles over 256 v-channels
  int mt = id & 3;     id >>= 2;   // 4 tiles over 64 k-channels
  int b = id;
  const int lane = threadIdx.x;
  const bf16_t* kbase = Kb + (long)b * 64 * HW + (long)(mt * 16) * HW;
  const bf16_t* vbase = V + (long)b * 256 * HW + (long)(jt * 16 + (lane & 15)) * HW;
  const float* ik = invk + (long)b * HW;
  const int kb2 = (lane >> 4) << 4;

  v8f acc = {};
  const int nend = chunk * 2048 + 2048;
  for (int nk = chunk * 2048; nk < nend; nk += 32) {
    v16bf a = a_frag_bf(kbase + nk, HW, lane);
    FragBF f;
    const bf16_t* vp = vbase + nk + kb2;
    const float*  ip = ik + nk + kb2;
#pragma unroll
    for (int t = 0; t < 16; ++t) f.h[t] = (bf16_t)((float)vp[t] * ip[t]);
    acc = wmma_bf16(a, f.v, acc);
  }
  float* sb = simT + (long)b * 256 * 64 + (long)(jt * 16 + (lane & 15)) * 64 + mt * 16;
  const int mh = (lane >> 4) << 3;
#pragma unroll
  for (int r = 0; r < 8; ++r)
    __hip_atomic_fetch_add(&sb[r + mh], acc[r], __ATOMIC_RELAXED,
                           __HIP_MEMORY_SCOPE_AGENT);
}

// ---------------- context^T[j][n] = sum_c simT[j][c] * q[c][n]*invq[n] ----------------
__global__ void k_ctx(const float* __restrict__ simT, const bf16_t* __restrict__ Q,
                      const float* __restrict__ invq, bf16_t* __restrict__ CTX) {
  __shared__ __align__(16) bf16_t bt[16 * BT_LD];
  const int lane = threadIdx.x & 31, wave = threadIdx.x >> 5, nw = blockDim.x >> 5;
  const int b = blockIdx.x >> 10;
  const int n0 = (blockIdx.x & 1023) << 4;
  const int m0 = (blockIdx.y * nw + wave) << 4;
  const bf16_t* qb = Q + (long)b * 64 * HW;
  const float* iq = invq + (long)b * HW;
  const float* ab = simT + (long)b * 256 * 64;

  v8f acc = {};
  for (int k0 = 0; k0 < 64; k0 += 32) {
    __syncthreads();
    for (int i = threadIdx.x; i < 256; i += blockDim.x) {
      int j = i & 15, kp = i >> 4;
      float s = iq[n0 + j];
      PackBF pk;
      pk.h[0] = (bf16_t)((float)qb[(long)(k0 + 2 * kp) * HW + n0 + j] * s);
      pk.h[1] = (bf16_t)((float)qb[(long)(k0 + 2 * kp + 1) * HW + n0 + j] * s);
      *(unsigned*)&bt[j * BT_LD + 2 * kp] = pk.u;
    }
    __syncthreads();
    v16bf a = a_frag_f32(ab + (long)m0 * 64 + k0, 64, lane);
    acc = wmma_bf16(a, b_frag_lds(bt, lane), acc);
  }
  const int n = n0 + (lane & 15);
  const int mh = (lane >> 4) << 3;
  bf16_t* cb = CTX + (long)b * 256 * HW;
#pragma unroll
  for (int r = 0; r < 8; ++r) cb[(long)(m0 + r + mh) * HW + n] = (bf16_t)acc[r];
}

// ---------- o-proj GEMM + BN + ReLU + residual + bilinear upsample-add -> FUSE ----------
__global__ void k_oproj(const bf16_t* __restrict__ CTX, const bf16_t* __restrict__ oW,
                        const float* __restrict__ osc, const float* __restrict__ osh,
                        const float* __restrict__ x, const float* __restrict__ up,
                        bf16_t* __restrict__ FUSE) {
  __shared__ __align__(16) bf16_t bt[16 * BT_LD];
  const int lane = threadIdx.x & 31, wave = threadIdx.x >> 5, nw = blockDim.x >> 5;
  const int b = blockIdx.x >> 10;
  const int n0 = (blockIdx.x & 1023) << 4;
  const int m0 = (blockIdx.y * nw + wave) << 4;
  const bf16_t* cb = CTX + (long)b * 256 * HW;

  v8f acc = {};
  for (int k0 = 0; k0 < 256; k0 += 32) {
    __syncthreads();
    for (int i = threadIdx.x; i < 256; i += blockDim.x) {
      int j = i & 15, kp = i >> 4;
      PackBF pk;
      pk.h[0] = cb[(long)(k0 + 2 * kp) * HW + n0 + j];
      pk.h[1] = cb[(long)(k0 + 2 * kp + 1) * HW + n0 + j];
      *(unsigned*)&bt[j * BT_LD + 2 * kp] = pk.u;
    }
    if (k0 < 224) __builtin_prefetch(cb + (long)(k0 + 32) * HW + n0, 0, 0);
    __syncthreads();
    v16bf a = a_frag_bf(oW + (long)m0 * 256 + k0, 256, lane);
    acc = wmma_bf16(a, b_frag_lds(bt, lane), acc);
  }
  const int n = n0 + (lane & 15);
  const int y = n >> 7, xp = n & 127;
  const int mh = (lane >> 4) << 3;
  const float* xb = x + (long)b * 256 * HW;
  bf16_t* fb = FUSE + (long)b * 256 * HW;
  // half-pixel bilinear coords for 64x64 -> 128x128 (align_corners=False)
  float sy = y * 0.5f - 0.25f, sx = xp * 0.5f - 0.25f;
  int y0 = (int)floorf(sy), x0 = (int)floorf(sx);
  float fy = sy - (float)y0, fx = sx - (float)x0;
  int y1 = min(63, max(0, y0 + 1)), x1 = min(63, max(0, x0 + 1));
  y0 = min(63, max(0, y0)); x0 = min(63, max(0, x0));
#pragma unroll
  for (int r = 0; r < 8; ++r) {
    int m = m0 + r + mh;
    float vv = fmaxf(acc[r] * osc[m] + osh[m], 0.f);   // conv1x1+BN+ReLU
    vv += xb[(long)m * HW + n];                        // residual
    const float* uc = up + (long)(b * 256 + m) * 4096;
    float a00 = uc[y0 * 64 + x0], a01 = uc[y0 * 64 + x1];
    float a10 = uc[y1 * 64 + x0], a11 = uc[y1 * 64 + x1];
    vv += a00 * (1.f - fy) * (1.f - fx) + a01 * (1.f - fy) * fx +
          a10 * fy * (1.f - fx) + a11 * fy * fx;       // upsample-add
    fb[(long)m * HW + n] = (bf16_t)vv;
  }
}

// ---------------- 3x3 conv as 9 shifted WMMA GEMMs + BN + ReLU ----------------
// Weights pre-reshaped to bf16 tap-major [r][128][256]: contiguous A-fragments.
__global__ void k_smooth(const bf16_t* __restrict__ FUSE, const bf16_t* __restrict__ Wsb,
                         const float* __restrict__ ssc, const float* __restrict__ ssh,
                         float* __restrict__ out) {
  __shared__ __align__(16) bf16_t bt[16 * BT_LD];
  const int lane = threadIdx.x & 31, wave = threadIdx.x >> 5;
  int id = blockIdx.x;
  const int xt = id & 7;   id >>= 3;   // 8 x-tiles of 16
  const int y  = id & 127; id >>= 7;   // 128 rows
  const int b  = id;                   // 4 batches
  const int x0 = xt << 4;
  const int o0 = wave << 4;            // 8 waves cover 128 out channels
  const bf16_t* fb = FUSE + (long)b * 256 * HW;

  v8f acc = {};
  for (int r = 0; r < 9; ++r) {
    const int dy = r / 3 - 1, dx = r % 3 - 1;
    const int yy = y + dy;
    const bool rowok = (yy >= 0) && (yy < 128);
    const bf16_t* wr = Wsb + (long)r * 128 * 256;
    for (int k0 = 0; k0 < 256; k0 += 32) {
      __syncthreads();
      for (int i = threadIdx.x; i < 256; i += blockDim.x) {
        int j = i & 15, kp = i >> 4;
        int xx = x0 + j + dx;
        PackBF pk; pk.u = 0u;
        if (rowok && xx >= 0 && xx < 128) {
          pk.h[0] = fb[(long)(k0 + 2 * kp) * HW + yy * 128 + xx];
          pk.h[1] = fb[(long)(k0 + 2 * kp + 1) * HW + yy * 128 + xx];
        }
        *(unsigned*)&bt[j * BT_LD + 2 * kp] = pk.u;
      }
      __syncthreads();
      v16bf a = a_frag_bf(wr + (long)o0 * 256 + k0, 256, lane);
      acc = wmma_bf16(a, b_frag_lds(bt, lane), acc);
    }
  }
  const int xp = x0 + (lane & 15);
  const int mh = (lane >> 4) << 3;
#pragma unroll
  for (int rr = 0; rr < 8; ++rr) {
    int o = o0 + rr + mh;
    out[(((long)b * 128 + o) * 128 + y) * 128 + xp] =
        fmaxf(acc[rr] * ssc[o] + ssh[o], 0.f);
  }
}

extern "C" void kernel_launch(void* const* d_in, const int* in_sizes, int n_in,
                              void* d_out, int out_size, void* d_ws, size_t ws_size,
                              hipStream_t stream) {
  (void)in_sizes; (void)n_in; (void)out_size; (void)ws_size;
  const float* x    = (const float*)d_in[0];
  const float* up   = (const float*)d_in[1];
  const float* qW   = (const float*)d_in[2];
  const float* q_sc = (const float*)d_in[3];
  const float* q_sh = (const float*)d_in[4];
  const float* kW   = (const float*)d_in[5];
  const float* k_sc = (const float*)d_in[6];
  const float* k_sh = (const float*)d_in[7];
  const float* vW   = (const float*)d_in[8];
  const float* v_sc = (const float*)d_in[9];
  const float* v_sh = (const float*)d_in[10];
  const float* oW   = (const float*)d_in[11];
  const float* o_sc = (const float*)d_in[12];
  const float* o_sh = (const float*)d_in[13];
  const float* sW   = (const float*)d_in[14];
  const float* s_sc = (const float*)d_in[15];
  const float* s_sh = (const float*)d_in[16];

  char* ws = (char*)d_ws;
  bf16_t* Q    = (bf16_t*)(ws);               //  8 MB  [B][64][HW]
  bf16_t* K    = (bf16_t*)(ws + 8388608);     //  8 MB  [B][64][HW]
  bf16_t* V    = (bf16_t*)(ws + 16777216);    // 32 MB  [B][256][HW]
  bf16_t* CTX  = (bf16_t*)(ws + 50331648);    // 32 MB  [B][256][HW]
  bf16_t* FUSE = (bf16_t*)(ws + 83886080);    // 32 MB  [B][256][HW]
  float* invq = (float*)(ws + 117440512);     // 256 KB [B][HW]
  float* invk = (float*)(ws + 117702656);     // 256 KB [B][HW]
  float* simT = (float*)(ws + 117964800);     // 256 KB [B][256][64]
  bf16_t* Wq  = (bf16_t*)(ws + 118226944);    // 32 KB
  bf16_t* Wk  = (bf16_t*)(ws + 118259712);    // 32 KB
  bf16_t* Wv  = (bf16_t*)(ws + 118292480);    // 128 KB
  bf16_t* Wo  = (bf16_t*)(ws + 118423552);    // 128 KB
  bf16_t* Wsb = (bf16_t*)(ws + 118554624);    // 576 KB [9][128][256]
  float* out = (float*)d_out;

  // weight pre-conversion (tiny, once per call)
  k_f2b<<<64, 256, 0, stream>>>(qW, Wq, 64 * 256);
  k_f2b<<<64, 256, 0, stream>>>(kW, Wk, 64 * 256);
  k_f2b<<<256, 256, 0, stream>>>(vW, Wv, 256 * 256);
  k_f2b<<<256, 256, 0, stream>>>(oW, Wo, 256 * 256);
  k_cvt_sw<<<1152, 256, 0, stream>>>(sW, Wsb);

  k_proj<<<dim3(4096, 1), 128, 0, stream>>>(x, Wq, q_sc, q_sh, Q, 64, 0);
  k_proj<<<dim3(4096, 1), 128, 0, stream>>>(x, Wk, k_sc, k_sh, K, 64, 0);
  k_proj<<<dim3(4096, 2), 256, 0, stream>>>(x, Wv, v_sc, v_sh, V, 256, 1);
  k_norm<<<256, 256, 0, stream>>>(Q, K, invq, invk);
  k_zero<<<256, 256, 0, stream>>>(simT, 65536);
  k_sim<<<2048, 32, 0, stream>>>(K, V, invk, simT);
  k_ctx<<<dim3(4096, 2), 256, 0, stream>>>(simT, Q, invq, CTX);
  k_oproj<<<dim3(4096, 2), 256, 0, stream>>>(CTX, Wo, o_sc, o_sh, x, up, FUSE);
  k_smooth<<<4096, 256, 0, stream>>>(FUSE, Wsb, s_sc, s_sh, out);
}
